// Attention_org_77300821394230
// MI455X (gfx1250) — matrix operand order repaired and checked by reading, not compiled, not run on hardware
//
#include <hip/hip_runtime.h>
#include <stdint.h>

// ---- problem constants (match reference) ----
#define B_ 8
#define N_ 4096
#define C_ 256
#define H_ 4

typedef __bf16 v16bf __attribute__((ext_vector_type(16)));
typedef float  v8f   __attribute__((ext_vector_type(8)));
typedef unsigned int u32x4 __attribute__((ext_vector_type(4)));

__device__ __forceinline__ uint16_t f2bf(float f) {
  uint32_t u = __float_as_uint(f);
  uint32_t r = u + 0x7FFFu + ((u >> 16) & 1u);   // round-to-nearest-even
  return (uint16_t)(r >> 16);
}

// low 32 bits of a generic (flat) shared pointer == wave-relative LDS byte offset
__device__ __forceinline__ uint32_t lds_off(const void* p) {
  return (uint32_t)(uintptr_t)p;
}

// async copy 16B global -> LDS, tracked by ASYNCcnt (CDNA5)
__device__ __forceinline__ void async_copy_b128(uint32_t lds_addr, const void* gaddr) {
  asm volatile("global_load_async_to_lds_b128 %0, %1, off"
               :: "v"(lds_addr), "v"(gaddr)
               : "memory");
}

__device__ __forceinline__ void wait_async0() {
  asm volatile("s_wait_asynccnt 0x0" ::: "memory");
}

// ---------------------------------------------------------------------------
// Convert emb1 fp32 -> bf16 row-major [B,N,C] and bf16 transposed [B,C,N]
// ---------------------------------------------------------------------------
__global__ void k_prep_emb(const float* __restrict__ emb,
                           uint16_t* __restrict__ emb_bf,
                           uint16_t* __restrict__ embT) {
  __shared__ uint16_t tile[32][33];
  int b  = blockIdx.z;
  int n0 = blockIdx.x * 32, c0 = blockIdx.y * 32;
  int tx = threadIdx.x, ty = threadIdx.y;
  uint16_t v = f2bf(emb[((size_t)b * N_ + n0 + ty) * C_ + c0 + tx]);
  emb_bf[((size_t)b * N_ + n0 + ty) * C_ + c0 + tx] = v;
  tile[ty][tx] = v;
  __syncthreads();
  embT[((size_t)b * C_ + c0 + ty) * N_ + n0 + tx] = tile[tx][ty];
}

// ---------------------------------------------------------------------------
// Convert weights: Wq,Wk -> bf16; Wv -> WvT2[c][h*256+d] = Wv[h][d][c]; Wout -> bf16
// ---------------------------------------------------------------------------
__global__ void k_prep_w(const float* __restrict__ Wq, const float* __restrict__ Wk,
                         const float* __restrict__ Wv, const float* __restrict__ Wout,
                         uint16_t* __restrict__ Wq_bf, uint16_t* __restrict__ Wk_bf,
                         uint16_t* __restrict__ WvT2, uint16_t* __restrict__ Wout_bf) {
  int idx = blockIdx.x * 256 + threadIdx.x;        // [0, H*C*C)
  int h = idx >> 16;
  int rem = idx & 65535;
  int d = rem >> 8;
  int c = rem & 255;
  Wq_bf[idx] = f2bf(Wq[idx]);
  Wk_bf[idx] = f2bf(Wk[idx]);
  WvT2[(size_t)c * (H_ * C_) + h * C_ + d] = f2bf(Wv[idx]);
  if (idx < C_ * C_) Wout_bf[idx] = f2bf(Wout[idx]);
}

// ---------------------------------------------------------------------------
// Generic bf16 WMMA GEMM:  D[m,n] = scale * sum_k A[m,k] * Bt[n,k]
// A: row-major (lda in halves), Bt: row-major (ldb), batched via (blockIdx.y, blockIdx.z)
// Block tile 128x64, 8 waves (4x2), each wave 32x32 (2x2 WMMA frags), K stage 64.
// Tiles staged via GLOBAL_LOAD_ASYNC_TO_LDS_B128 into double-buffered LDS:
// one workgroup barrier per K stage, copy of stage s+1 overlaps WMMAs of stage s.
// ---------------------------------------------------------------------------
#define BM 128
#define BN 64
#define BK 64
#define LDT 72   // LDS row stride in halves (144 bytes, 16B multiple)

__global__ __launch_bounds__(256) void k_gemm_bf16(
    const uint16_t* __restrict__ A, long sAy, long sAz, int lda,
    const uint16_t* __restrict__ Bt, long sBy, long sBz, int ldb,
    void* __restrict__ D, long sDy, long sDz, int ldd,
    int M, int N, int K, float scale, int out_f32, int transD) {
  __shared__ __align__(16) uint16_t As[2][BM * LDT];
  __shared__ __align__(16) uint16_t Bs[2][BN * LDT];

  const uint16_t* Ab = A  + (size_t)blockIdx.y * sAy + (size_t)blockIdx.z * sAz;
  const uint16_t* Bb = Bt + (size_t)blockIdx.y * sBy + (size_t)blockIdx.z * sBz;

  int tiles_n = N / BN;
  int bm0 = (blockIdx.x / tiles_n) * BM;
  int bn0 = (blockIdx.x % tiles_n) * BN;

  int tid  = threadIdx.x;
  int lane = tid & 31, wid = tid >> 5;
  int wm = (wid & 3) * 32;    // 4 waves along M
  int wn = (wid >> 2) * 32;   // 2 waves along N

  // per-thread fixed chunk assignment (16B = 8 halves per chunk)
  // A: 128 rows x 8 chunks -> 4 chunks/thread ; B: 64 rows x 8 chunks -> 2 chunks/thread
  auto issue_stage = [&](int k0, int p) {
#pragma unroll
    for (int i = 0; i < 4; ++i) {
      int c = tid + i * 256;
      int r = c >> 3, cc = c & 7;
      async_copy_b128(lds_off(&As[p][r * LDT + cc * 8]),
                      &Ab[(size_t)(bm0 + r) * lda + k0 + cc * 8]);
    }
#pragma unroll
    for (int i = 0; i < 2; ++i) {
      int c = tid + i * 256;
      int r = c >> 3, cc = c & 7;
      async_copy_b128(lds_off(&Bs[p][r * LDT + cc * 8]),
                      &Bb[(size_t)(bn0 + r) * ldb + k0 + cc * 8]);
    }
  };

  v8f acc[2][2] = {};

  int nst = K / BK;
  issue_stage(0, 0);
  wait_async0();
  __syncthreads();                    // buffer 0 ready for the whole workgroup

  for (int s = 0; s < nst; ++s) {
    int p = s & 1;
    if (s + 1 < nst) issue_stage((s + 1) * BK, p ^ 1);   // overlap copy with compute

#pragma unroll
    for (int kc = 0; kc < 2; ++kc) {
      int kb = kc * 32;
      union Frag { struct { u32x4 lo, hi; } q; v16bf v; } a[2], b[2];
      // A fragment (16x32 bf16): lane<16 -> K 0..7 & 16..23 ; lane>=16 -> K 8..15 & 24..31
      int arow = lane & 15;
      int aoff = kb + ((lane < 16) ? 0 : 8);
#pragma unroll
      for (int i = 0; i < 2; ++i) {
        const uint16_t* ap = &As[p][(wm + i * 16 + arow) * LDT + aoff];
        a[i].q.lo = *(const u32x4*)ap;
        a[i].q.hi = *(const u32x4*)(ap + 16);
      }
      // B fragment (32x16 bf16, from Bt rows): lane<16 -> K 0..7 & 16..23 ; lane>=16 -> K 8..15 & 24..31
      int boff = kb + ((lane < 16) ? 0 : 8);
#pragma unroll
      for (int j = 0; j < 2; ++j) {
        const uint16_t* bp = &Bs[p][(wn + j * 16 + (lane & 15)) * LDT + boff];
        b[j].q.lo = *(const u32x4*)bp;
        b[j].q.hi = *(const u32x4*)(bp + 16);
      }
#pragma unroll
      for (int i = 0; i < 2; ++i)
#pragma unroll
        for (int j = 0; j < 2; ++j)
          acc[i][j] = __builtin_amdgcn_wmma_f32_16x16x32_bf16(
              false, a[i].v, false, b[j].v, (short)0, acc[i][j], false, false);
    }

    wait_async0();                     // stage s+1 copies have landed in LDS
    __syncthreads();                   // everyone done reading buffer p; s+1 buffer visible
  }

  // epilogue: C/D layout: VGPR r -> (M=r, N=lane) lanes 0-15 ; (M=r+8, N=lane-16) lanes 16-31
  char* Db = (char*)D +
             ((size_t)blockIdx.y * sDy + (size_t)blockIdx.z * sDz) * (out_f32 ? 4 : 2);
#pragma unroll
  for (int i = 0; i < 2; ++i)
#pragma unroll
    for (int j = 0; j < 2; ++j) {
      int mb = bm0 + wm + i * 16 + ((lane < 16) ? 0 : 8);
      int nb = bn0 + wn + j * 16 + (lane & 15);
#pragma unroll
      for (int r = 0; r < 8; ++r) {
        float val = acc[i][j][r] * scale;
        int mm = mb + r, nn = nb;
        size_t off = transD ? ((size_t)nn * ldd + mm) : ((size_t)mm * ldd + nn);
        if (out_f32) ((float*)Db)[off] = val;
        else         ((uint16_t*)Db)[off] = f2bf(val);
      }
    }
  (void)M;
}

// ---------------------------------------------------------------------------
// InstanceNorm over CxC per (b,h) + row softmax.
// probsT2 (bf16) layout [b][i][h][j] so the M-GEMM sees contiguous K=H*C rows.
// probsF (fp32) same layout, used for the `weights` head-mean output.
// ---------------------------------------------------------------------------
__global__ __launch_bounds__(256) void k_norm_softmax(
    const float* __restrict__ Sf, uint16_t* __restrict__ probsT2,
    float* __restrict__ probsF) {
  int b = blockIdx.x >> 2, h = blockIdx.x & 3;
  const float* S = Sf + (size_t)(b * H_ + h) * (C_ * C_);
  int tid = threadIdx.x;
  __shared__ float red[512];
  __shared__ float stats[2];

  float s = 0.f, ss = 0.f;
  for (int i = tid; i < C_ * C_; i += 256) { float x = S[i]; s += x; ss += x * x; }
  red[tid] = s; red[256 + tid] = ss;
  __syncthreads();
  for (int o = 128; o > 0; o >>= 1) {
    if (tid < o) { red[tid] += red[tid + o]; red[256 + tid] += red[256 + tid + o]; }
    __syncthreads();
  }
  if (tid == 0) {
    float mu  = red[0] * (1.f / 65536.f);
    float var = red[256] * (1.f / 65536.f) - mu * mu;   // biased variance
    stats[0] = mu;
    stats[1] = rsqrtf(var + 1e-5f);
  }
  __syncthreads();
  float mu = stats[0], rstd = stats[1];

  int lane = tid & 31, wid = tid >> 5;
  for (int row = wid; row < C_; row += 8) {
    const float* Sr = S + (size_t)row * C_;
    float v[8], mx = -1e30f;
#pragma unroll
    for (int t = 0; t < 8; ++t) { v[t] = (Sr[lane + t * 32] - mu) * rstd; mx = fmaxf(mx, v[t]); }
#pragma unroll
    for (int o = 16; o > 0; o >>= 1) mx = fmaxf(mx, __shfl_xor(mx, o, 32));
    float sum = 0.f;
#pragma unroll
    for (int t = 0; t < 8; ++t) { v[t] = __expf(v[t] - mx); sum += v[t]; }
#pragma unroll
    for (int o = 16; o > 0; o >>= 1) sum += __shfl_xor(sum, o, 32);
    float inv = 1.f / sum;
    size_t base = (((size_t)(b * C_ + row)) * H_ + h) * C_;
#pragma unroll
    for (int t = 0; t < 8; ++t) {
      float p = v[t] * inv;
      probsT2[base + lane + t * 32] = f2bf(p);
      probsF[base + lane + t * 32]  = p;
    }
  }
}

// weights[b,i,j] = mean_h probs[b,h,i,j]   (fp32, second output)
__global__ void k_weights(const float* __restrict__ probsF, float* __restrict__ outw) {
  int idx = blockIdx.x * 256 + threadIdx.x;   // [0, B*C*C)
  int j = idx & 255;
  int bi = idx >> 8;                          // b*C + i
  const float* p = probsF + (size_t)bi * (H_ * C_) + j;
  outw[idx] = 0.25f * (p[0] + p[C_] + p[2 * C_] + p[3 * C_]);
}

// ---------------------------------------------------------------------------
extern "C" void kernel_launch(void* const* d_in, const int* in_sizes, int n_in,
                              void* d_out, int out_size, void* d_ws, size_t ws_size,
                              hipStream_t stream) {
  (void)in_sizes; (void)n_in; (void)out_size; (void)ws_size;
  const float* emb  = (const float*)d_in[0];
  const float* Wq   = (const float*)d_in[1];
  const float* Wk   = (const float*)d_in[2];
  const float* Wv   = (const float*)d_in[3];
  const float* Wout = (const float*)d_in[4];

  char* ws = (char*)d_ws;
  size_t o = 0;
  uint16_t* emb_bf  = (uint16_t*)(ws + o); o += (size_t)B_ * N_ * C_ * 2;       // 16 MB
  uint16_t* embT    = (uint16_t*)(ws + o); o += (size_t)B_ * C_ * N_ * 2;       // 16 MB
  uint16_t* Wq_bf   = (uint16_t*)(ws + o); o += (size_t)H_ * C_ * C_ * 2;
  uint16_t* Wk_bf   = (uint16_t*)(ws + o); o += (size_t)H_ * C_ * C_ * 2;
  uint16_t* WvT2    = (uint16_t*)(ws + o); o += (size_t)C_ * H_ * C_ * 2;
  uint16_t* Wout_bf = (uint16_t*)(ws + o); o += (size_t)C_ * C_ * 2;
  uint16_t* G_bf    = (uint16_t*)(ws + o); o += (size_t)B_ * C_ * C_ * 2;
  uint16_t* U_bf    = (uint16_t*)(ws + o); o += (size_t)B_ * H_ * C_ * C_ * 2;
  float*    Sf      = (float*)(ws + o);    o += (size_t)B_ * H_ * C_ * C_ * 4;
  uint16_t* probsT2 = (uint16_t*)(ws + o); o += (size_t)B_ * H_ * C_ * C_ * 2;
  float*    probsF  = (float*)(ws + o);    o += (size_t)B_ * H_ * C_ * C_ * 4;
  uint16_t* M_bf    = (uint16_t*)(ws + o); o += (size_t)B_ * C_ * C_ * 2;
  uint16_t* Pt_bf   = (uint16_t*)(ws + o); o += (size_t)B_ * C_ * C_ * 2;       // ~64 MB total

  float* O1   = (float*)d_out;
  float* outw = O1 + (size_t)B_ * N_ * C_;

  // 0) conversions / transposes
  k_prep_emb<<<dim3(N_ / 32, C_ / 32, B_), dim3(32, 32), 0, stream>>>(emb, emb_bf, embT);
  k_prep_w<<<dim3((H_ * C_ * C_) / 256), dim3(256), 0, stream>>>(
      Wq, Wk, Wv, Wout, Wq_bf, Wk_bf, WvT2, Wout_bf);

  // 1) G[b] = emb1[b]^T emb1[b]   (M=N=256, K=4096)  -- symmetric Gram matrix
  k_gemm_bf16<<<dim3(8, B_, 1), 256, 0, stream>>>(
      embT, (long)C_ * N_, 0, N_,  embT, (long)C_ * N_, 0, N_,
      G_bf, (long)C_ * C_, 0, C_,  256, 256, 4096, 1.f, 0, 0);

  // 2a) U[b,h] = Wq[h] @ G[b]   (G symmetric -> Bt = G row-major)
  k_gemm_bf16<<<dim3(8, H_, B_), 256, 0, stream>>>(
      Wq_bf, (long)C_ * C_, 0, C_,  G_bf, 0, (long)C_ * C_, C_,
      U_bf, (long)C_ * C_, (long)H_ * C_ * C_, C_,  256, 256, 256, 1.f, 0, 0);

  // 2b) S[b,h] = U[b,h] @ Wk[h]^T  (scale 1/sqrt(C) cancels in instance-norm)
  k_gemm_bf16<<<dim3(8, H_, B_), 256, 0, stream>>>(
      U_bf, (long)C_ * C_, (long)H_ * C_ * C_, C_,  Wk_bf, (long)C_ * C_, 0, C_,
      Sf, (long)C_ * C_, (long)H_ * C_ * C_, C_,  256, 256, 256, 1.f, 1, 0);

  // 3) instance-norm + softmax -> probs (bf16 [b][i][h][j]) + fp32 copy
  k_norm_softmax<<<dim3(B_ * H_), dim3(256), 0, stream>>>(Sf, probsT2, probsF);

  // 4a) M[b][c][i] = (1/H) sum_{h,j} Wv[h,j,c] * probs[b,h,i,j]   (K = H*C = 1024)
  k_gemm_bf16<<<dim3(8, B_, 1), 256, 0, stream>>>(
      WvT2, 0, 0, H_ * C_,  probsT2, (long)C_ * H_ * C_, 0, H_ * C_,
      M_bf, (long)C_ * C_, 0, C_,  256, 256, H_ * C_, 0.25f, 0, 0);

  // 4b) P[b] = M[b] @ Wout^T, stored transposed (Pt[d][c]) for step 5
  k_gemm_bf16<<<dim3(8, B_, 1), 256, 0, stream>>>(
      M_bf, (long)C_ * C_, 0, C_,  Wout_bf, 0, 0, C_,
      Pt_bf, (long)C_ * C_, 0, C_,  256, 256, 256, 1.f, 0, 1);

  // 5) O1[b] = emb1[b] @ P[b]   (M=4096, N=256, K=256) -> fp32 output
  k_gemm_bf16<<<dim3((N_ / BM) * (C_ / BN), B_, 1), 256, 0, stream>>>(
      emb_bf, (long)N_ * C_, 0, C_,  Pt_bf, (long)C_ * C_, 0, C_,
      O1, (long)N_ * C_, 0, C_,  N_, 256, 256, 1.f, 1, 0);

  // 6) weights = mean over heads
  k_weights<<<dim3((B_ * C_ * C_) / 256), dim3(256), 0, stream>>>(probsF, outw);
}
